// MultiBoxLoss_26946624815860
// MI455X (gfx1250) — compile-verified, matrix operand order checked by reading, not run on hardware
//
#include <hip/hip_runtime.h>

#define BB 32
#define PP 24656
#define NC 81
#define OO 16
#define ROW 85                // 4 loc + 81 conf floats per prior
#define PT (PP / 16)          // 1541 tiles of 16 priors per batch
#define NTILES (BB * PT)      // 49312
#define NBINS 1024
#define TILE_DW (16 * ROW)    // 1360 dwords per tile
#define TILE_BYTES (TILE_DW * 4)   // 5440
#define TILE_VEC4 (TILE_BYTES / 16) // 340 b128 transfers per tile

typedef __attribute__((ext_vector_type(16))) _Float16 v16h;
typedef __attribute__((ext_vector_type(8)))  float    v8f;

// ---------------- workspace layout (bytes) ----------------
#define OFF_GBEST   0                           // u64[BB*OO]
#define OFF_ACC     4096                        // f32 acc[2]: loss_l, loss_c
#define OFF_NUMPOS  4104                        // i32[BB]
#define OFF_TAILCNT 4232                        // i32[BB]
#define OFF_TBIN    4360                        // i32[BB]
#define OFF_REM     4488                        // i32[BB]
#define OFF_HCNT    5120                        // u32[BB*NBINS]
#define OFF_HSUM    (5120 + 4*BB*NBINS)         // f32[BB*NBINS]
#define OFF_BTOV    (OFF_HSUM + 4*BB*NBINS)     // f32[BB*PP]
#define OFF_BTIDX   (OFF_BTOV  + 4*(size_t)BB*PP)
#define OFF_CE      (OFF_BTIDX + 4*(size_t)BB*PP)
#define OFF_CONF    (OFF_CE    + 4*(size_t)BB*PP)
#define ZERO_WORDS  (OFF_BTOV / 4)              // zero gbest+scalars+hists

__global__ void k_zero(unsigned* w, int nwords) {
    int i = blockIdx.x * blockDim.x + threadIdx.x;
    if (i < nwords) w[i] = 0u;
}

// One thread per (batch, prior): IoU vs 16 truths; best-truth per prior to ws;
// best-prior per truth via packed u64 atomicMax (LDS then global).
__global__ void k_match(const float* __restrict__ tboxes, const float* __restrict__ priors,
                        float* __restrict__ bt_ov, int* __restrict__ bt_idx,
                        unsigned long long* __restrict__ gbest) {
    __shared__ unsigned long long sb[OO];
    int tid = threadIdx.x;
    int b = blockIdx.y;
    int p = blockIdx.x * blockDim.x + tid;
    if (tid < OO) sb[tid] = 0ull;
    __syncthreads();
    if (p < PP) {
        float cx = priors[p*4+0], cy = priors[p*4+1], pw = priors[p*4+2], ph = priors[p*4+3];
        float bx0 = cx - 0.5f*pw, by0 = cy - 0.5f*ph, bx1 = cx + 0.5f*pw, by1 = cy + 0.5f*ph;
        float areaB = pw * ph;
        float bestov = -1.0f; int bestti = 0;
        for (int ti = 0; ti < OO; ++ti) {
            const float* t = tboxes + ((size_t)b*OO + ti)*4;
            float ax0 = t[0], ay0 = t[1], ax1 = t[2], ay1 = t[3];
            float lx = fmaxf(ax0, bx0), ly = fmaxf(ay0, by0);
            float rx = fminf(ax1, bx1), ry = fminf(ay1, by1);
            float iw = fmaxf(rx - lx, 0.0f), ih = fmaxf(ry - ly, 0.0f);
            float inter = iw * ih;
            float areaA = (ax1 - ax0) * (ay1 - ay0);
            float ov = inter / (areaA + areaB - inter);
            if (ov > bestov) { bestov = ov; bestti = ti; }   // first-max tie-break
            // pack: larger iou wins; among ties, smallest prior index (argmax semantics)
            unsigned long long pk = ((unsigned long long)__float_as_uint(ov) << 32)
                                  | (unsigned)(~(unsigned)p);
            atomicMax(&sb[ti], pk);
        }
        bt_ov [(size_t)b*PP + p] = bestov;
        bt_idx[(size_t)b*PP + p] = bestti;
    }
    __syncthreads();
    if (tid < OO) atomicMax(&gbest[b*OO + tid], sb[tid]);
}

// Force-match: best prior of each truth gets ov=2.0 and idx=o (sequential per batch).
__global__ void k_override(const unsigned long long* __restrict__ gbest,
                           float* __restrict__ bt_ov, int* __restrict__ bt_idx) {
    int b = threadIdx.x;
    if (b >= BB) return;
    for (int o = 0; o < OO; ++o) {
        unsigned j = ~(unsigned)(gbest[b*OO + o] & 0xFFFFFFFFull);
        if (j < PP) { bt_ov[(size_t)b*PP + j] = 2.0f; bt_idx[(size_t)b*PP + j] = o; }
    }
}

// Main kernel: one wave per 16-prior tile.
//  1) stage the tile's 16x85 f32 rows into LDS with GLOBAL_LOAD_ASYNC_TO_LDS_B128
//     (coalesced DMA, ASYNCcnt), then s_wait_asynccnt 0.
//  2) per-prior max over classes from LDS (split across half-waves).
//  3) sum-of-exp via 3x V_WMMA_F32_16X16X32_F16 against an all-ones B (rowsum-as-matmul).
//  4) CE, smooth-L1 loc loss for positives, histogram binning for negatives.
__global__ void __launch_bounds__(256) k_main(
        const float* __restrict__ pred, const float* __restrict__ tboxes,
        const int* __restrict__ tlabels, const float* __restrict__ priors,
        const float* __restrict__ bt_ov, const int* __restrict__ bt_idx,
        float* __restrict__ ce_out, int* __restrict__ conf_out,
        float* __restrict__ acc, int* __restrict__ num_pos,
        unsigned* __restrict__ hcnt, float* __restrict__ hsum) {
    __shared__ __align__(16) float smem[8 * TILE_DW];   // 43.5 KB: 8 tiles
    int lane = threadIdx.x & 31;
    int wave = threadIdx.x >> 5;
    int t  = blockIdx.x * 8 + wave;        // tile id, exact: NTILES = gridDim*8
    int b  = t / PT;
    int p0 = (t - b * PT) * 16;
    int m  = lane & 15;                    // prior-in-tile (M row)
    int hf = lane >> 4;                    // half-wave id
    int p  = p0 + m;
    size_t bp = (size_t)b*PP + p;

    // ---- async DMA: tile rows (contiguous 5440 B in global) -> LDS ----
    float* sm = smem + wave * TILE_DW;
    unsigned ldsbase = (unsigned)(unsigned long long)(void*)sm;        // LDS byte offset
    unsigned gtile   = (unsigned)(((size_t)b*PP + (size_t)p0) * (ROW*4)); // byte offset, < 2^31
    unsigned long long sbase = (unsigned long long)(const void*)pred;
    #pragma unroll
    for (int i = 0; i < (TILE_VEC4 + 31) / 32; ++i) {
        int idx = i * 32 + lane;                // b128 transfer index
        if (idx < TILE_VEC4) {
            unsigned ldsoff = ldsbase + (unsigned)idx * 16u;
            unsigned voff   = gtile   + (unsigned)idx * 16u;
            asm volatile("global_load_async_to_lds_b128 %0, %1, %2"
                         :: "v"(ldsoff), "v"(voff), "s"(sbase) : "memory");
        }
    }
    asm volatile("s_wait_asynccnt 0x0" ::: "memory");

    float ov = bt_ov[bp];
    int   ti = bt_idx[bp];
    int conft = (ov < 0.5f) ? 0 : (tlabels[b*OO + ti] + 1);
    int lbl = conft > (NC - 1) ? (NC - 1) : conft;
    bool pos = conft > 0;

    // per-prior max over 81 classes (LDS): half 0 even classes, half 1 odd, combine.
    const float* row = sm + m * ROW;            // [0..3]=loc, [4..84]=conf
    float mx = -3.0e38f;
    for (int c = hf; c < NC; c += 2) mx = fmaxf(mx, row[4 + c]);
    mx = fmaxf(mx, __shfl_xor(mx, 16, 32));
    float xl = row[4 + lbl];

    // sumexp = (exp(conf - mx)) x ones via WMMA. 16-bit A layout (ISA 7.12.2):
    // lane<16: elems 0-7 -> K 0-7, elems 8-15 -> K 16-23; lane>=16: +8.
    v16h ones;
    #pragma unroll
    for (int e = 0; e < 16; ++e) ones[e] = (_Float16)1.0f;
    v8f accv = {};
    #pragma unroll
    for (int chunk = 0; chunk < 3; ++chunk) {
        int kbase = chunk * 32;
        v16h a;
        #pragma unroll
        for (int e = 0; e < 16; ++e) {
            int k = (e & 7) + ((e >> 3) << 4) + hf * 8;
            int c = kbase + k;
            float val = (c < NC) ? __expf(row[4 + c] - mx) : 0.0f;   // <= 1.0, f16-safe
            a[e] = (_Float16)val;
        }
        accv = __builtin_amdgcn_wmma_f32_16x16x32_f16(
                   false, a, false, ones, (short)0, accv, false, false);
    }
    // D[m][n] = sumexp(prior m) for all n; lanes 0-15 hold M 0-7, lanes 16-31 hold M 8-15.
    float v = 0.0f;
    #pragma unroll
    for (int r = 0; r < 8; ++r) v = ((m & 7) == r) ? accv[r] : v;
    float pv = __shfl_xor(v, 16, 32);
    float s = ((m < 8) == (hf == 0)) ? v : pv;

    float ce = fmaxf(mx + __logf(s) - xl, 0.0f);

    if (hf == 0) {
        ce_out[bp]   = ce;
        conf_out[bp] = conft;
        unsigned long long bal = __ballot(pos);
        if (lane == 0) {
            int np = __popcll(bal);
            if (np) atomicAdd(&num_pos[b], np);
        }
        if (pos) {
            atomicAdd(&acc[1], ce);                       // sel includes positives
            const float* tr = tboxes + ((size_t)b*OO + ti)*4;
            float pcx = priors[p*4+0], pcy = priors[p*4+1];
            float pw  = priors[p*4+2], ph  = priors[p*4+3];
            float gx = (0.5f*(tr[0]+tr[2]) - pcx) / (0.1f * pw);
            float gy = (0.5f*(tr[1]+tr[3]) - pcy) / (0.1f * ph);
            float gw = __logf((tr[2]-tr[0]) / pw) * 5.0f;  // /0.2
            float gh = __logf((tr[3]-tr[1]) / ph) * 5.0f;
            float lt[4] = {gx, gy, gw, gh};
            float ll = 0.0f;
            #pragma unroll
            for (int k = 0; k < 4; ++k) {
                float d = row[k] - lt[k];                  // loc from staged LDS row
                float ad = fabsf(d);
                ll += (ad < 1.0f) ? 0.5f*d*d : (ad - 0.5f);
            }
            atomicAdd(&acc[0], ll);
        } else {
            unsigned bin = (__float_as_uint(ce) >> 21) & (NBINS - 1);
            atomicAdd(&hcnt[b*NBINS + bin], 1u);
            atomicAdd(&hsum[b*NBINS + bin], ce);
        }
    }
}

// Per-batch: scan histogram top-down to find threshold bin for top-(3*num_pos).
__global__ void k_select(const unsigned* __restrict__ hcnt, const float* __restrict__ hsum,
                         const int* __restrict__ num_pos, float* __restrict__ acc,
                         int* __restrict__ Tbin, int* __restrict__ remv,
                         int* __restrict__ tailcnt) {
    int b = blockIdx.x;
    if (threadIdx.x != 0) return;
    long long nn = (long long)num_pos[b] * 3;
    if (nn > PP) nn = PP;
    long long cum = 0; float lsum = 0.0f; int T = -1; int rem = 0;
    for (int bin = NBINS - 1; bin >= 0; --bin) {
        unsigned c = hcnt[b*NBINS + bin];
        if ((long long)c + cum >= nn) { T = bin; rem = (int)(nn - cum); break; }
        cum  += c;
        lsum += hsum[b*NBINS + bin];
    }
    if (lsum != 0.0f) atomicAdd(&acc[1], lsum);
    Tbin[b] = T; remv[b] = rem; tailcnt[b] = 0;
}

// Admit `rem` elements from the threshold bin.
__global__ void k_tail(const float* __restrict__ ce, const int* __restrict__ conf_t,
                       const int* __restrict__ Tbin, const int* __restrict__ remv,
                       int* __restrict__ tailcnt, float* __restrict__ acc) {
    int b = blockIdx.y;
    int p = blockIdx.x * blockDim.x + threadIdx.x;
    if (p >= PP) return;
    int T = Tbin[b];
    if (T < 0) return;                       // everything already summed
    size_t bp = (size_t)b*PP + p;
    if (conf_t[bp] > 0) return;
    float c = ce[bp];
    if ((int)((__float_as_uint(c) >> 21) & (NBINS - 1)) != T) return;
    int slot = atomicAdd(&tailcnt[b], 1);
    if (slot < remv[b]) atomicAdd(&acc[1], c);
}

__global__ void k_final(const float* __restrict__ acc, const int* __restrict__ num_pos,
                        float* __restrict__ out) {
    if (threadIdx.x != 0) return;
    int N = 0;
    for (int b = 0; b < BB; ++b) N += num_pos[b];
    float fn = (float)N;
    out[0] = acc[0] / fn;
    out[1] = acc[1] / fn;
}

extern "C" void kernel_launch(void* const* d_in, const int* in_sizes, int n_in,
                              void* d_out, int out_size, void* d_ws, size_t ws_size,
                              hipStream_t stream) {
    const float* pred    = (const float*)d_in[0];   // (B,P,85)
    const float* tboxes  = (const float*)d_in[1];   // (B,O,4)
    const int*   tlabels = (const int*)  d_in[2];   // (B,O)
    const float* priors  = (const float*)d_in[3];   // (P,4)
    float* out = (float*)d_out;

    char* ws = (char*)d_ws;
    unsigned long long* gbest = (unsigned long long*)(ws + OFF_GBEST);
    float*    acc     = (float*)   (ws + OFF_ACC);
    int*      num_pos = (int*)     (ws + OFF_NUMPOS);
    int*      tailcnt = (int*)     (ws + OFF_TAILCNT);
    int*      Tbin    = (int*)     (ws + OFF_TBIN);
    int*      remv    = (int*)     (ws + OFF_REM);
    unsigned* hcnt    = (unsigned*)(ws + OFF_HCNT);
    float*    hsum    = (float*)   (ws + OFF_HSUM);
    float*    bt_ov   = (float*)   (ws + OFF_BTOV);
    int*      bt_idx  = (int*)     (ws + OFF_BTIDX);
    float*    ce      = (float*)   (ws + OFF_CE);
    int*      conf_t  = (int*)     (ws + OFF_CONF);

    k_zero<<<(ZERO_WORDS + 255) / 256, 256, 0, stream>>>((unsigned*)ws, (int)ZERO_WORDS);

    dim3 gm((PP + 255) / 256, BB);
    k_match<<<gm, 256, 0, stream>>>(tboxes, priors, bt_ov, bt_idx, gbest);
    k_override<<<1, BB, 0, stream>>>(gbest, bt_ov, bt_idx);

    k_main<<<NTILES / 8, 256, 0, stream>>>(pred, tboxes, tlabels, priors,
                                           bt_ov, bt_idx, ce, conf_t,
                                           acc, num_pos, hcnt, hsum);

    k_select<<<BB, 32, 0, stream>>>(hcnt, hsum, num_pos, acc, Tbin, remv, tailcnt);

    dim3 gt((PP + 255) / 256, BB);
    k_tail<<<gt, 256, 0, stream>>>(ce, conf_t, Tbin, remv, tailcnt, acc);

    k_final<<<1, 1, 0, stream>>>(acc, num_pos, out);
}
